// Attention_62242666054124
// MI455X (gfx1250) — compile-verified, matrix operand order checked by reading, not run on hardware
//
#include <hip/hip_runtime.h>
#include <hip/hip_bf16.h>
#include <math.h>

typedef __attribute__((ext_vector_type(16))) __bf16 v16bf;
typedef __attribute__((ext_vector_type(8)))  __bf16 v8bf;
typedef __attribute__((ext_vector_type(8)))  float  v8f;

#define BN 1024
#define KN 2048
#define HN 128
#define DN 512

// ---------------- elementwise f32 -> bf16 convert ----------------
__global__ void cvt_f32_bf16(const float* __restrict__ in, __bf16* __restrict__ out, int n) {
    int i = blockIdx.x * blockDim.x + threadIdx.x;
    if (i < n) out[i] = (__bf16)in[i];
}

// ------------- values [KN,DN] f32 -> vT [DN,KN] bf16 -------------
__global__ void cvt_transpose_vals(const float* __restrict__ in, __bf16* __restrict__ out) {
    __shared__ float tile[32][33];
    const int k0 = blockIdx.x * 32;
    const int n0 = blockIdx.y * 32;
    const int tx = threadIdx.x & 31;
    const int ty = threadIdx.x >> 5;           // 0..7
    #pragma unroll
    for (int r = ty; r < 32; r += 8)
        tile[r][tx] = in[(size_t)(k0 + r) * DN + n0 + tx];
    __syncthreads();
    #pragma unroll
    for (int r = ty; r < 32; r += 8)
        out[(size_t)(n0 + r) * KN + k0 + tx] = (__bf16)tile[tx][r];
}

// ------- WMMA bf16 GEMM: C[M,N] = A[M,Kd] * Bm[N,Kd]^T, 32x32 tile/wave -------
// Register-blocked: 4 f32 accumulators; each A/B fragment feeds 2 WMMAs
// (2 b128 loads per WMMA from L2-resident operands).
// A-layout (16-bit A 16x32, wave32): lanes 0-15 = row M=lane, K {0..7,16..23};
// lanes 16-31 = row M=lane-16, K {8..15,24..31}. B symmetric (column = Bm row).
__global__ void gemm_bf16_nt(const __bf16* __restrict__ A, const __bf16* __restrict__ Bm,
                             float* __restrict__ C, int M, int N, int Kd) {
    const int lane   = threadIdx.x & 31;
    const int wave   = threadIdx.x >> 5;
    const int tilesN = N >> 5;                 // 32-wide column tiles
    const int tile   = blockIdx.x * (blockDim.x >> 5) + wave;
    if (tile >= (M >> 5) * tilesN) return;
    const int tm  = (tile / tilesN) << 5;
    const int tn  = (tile % tilesN) << 5;
    const int r16 = lane & 15;
    const int hi  = lane >> 4;

    const __bf16* aRow0 = A  + (size_t)(tm + r16) * Kd + hi * 8;
    const __bf16* aRow1 = aRow0 + (size_t)16 * Kd;
    const __bf16* bRow0 = Bm + (size_t)(tn + r16) * Kd + hi * 8;
    const __bf16* bRow1 = bRow0 + (size_t)16 * Kd;

    v8f acc00 = {}, acc01 = {}, acc10 = {}, acc11 = {};
    for (int k0 = 0; k0 < Kd; k0 += 32) {
        v8bf a0l = *(const v8bf*)(aRow0 + k0);
        v8bf a0h = *(const v8bf*)(aRow0 + k0 + 16);
        v8bf a1l = *(const v8bf*)(aRow1 + k0);
        v8bf a1h = *(const v8bf*)(aRow1 + k0 + 16);
        v8bf b0l = *(const v8bf*)(bRow0 + k0);
        v8bf b0h = *(const v8bf*)(bRow0 + k0 + 16);
        v8bf b1l = *(const v8bf*)(bRow1 + k0);
        v8bf b1h = *(const v8bf*)(bRow1 + k0 + 16);
        v16bf av0 = __builtin_shufflevector(a0l, a0h, 0,1,2,3,4,5,6,7,8,9,10,11,12,13,14,15);
        v16bf av1 = __builtin_shufflevector(a1l, a1h, 0,1,2,3,4,5,6,7,8,9,10,11,12,13,14,15);
        v16bf bv0 = __builtin_shufflevector(b0l, b0h, 0,1,2,3,4,5,6,7,8,9,10,11,12,13,14,15);
        v16bf bv1 = __builtin_shufflevector(b1l, b1h, 0,1,2,3,4,5,6,7,8,9,10,11,12,13,14,15);
        acc00 = __builtin_amdgcn_wmma_f32_16x16x32_bf16(false, av0, false, bv0, (short)0, acc00, false, false);
        acc01 = __builtin_amdgcn_wmma_f32_16x16x32_bf16(false, av0, false, bv1, (short)0, acc01, false, false);
        acc10 = __builtin_amdgcn_wmma_f32_16x16x32_bf16(false, av1, false, bv0, (short)0, acc10, false, false);
        acc11 = __builtin_amdgcn_wmma_f32_16x16x32_bf16(false, av1, false, bv1, (short)0, acc11, false, false);
    }

    // C/D layout: VGPR r, lanes 0-15 -> M=r, lanes 16-31 -> M=8+r; N = lane&15.
    float* c00 = C + (size_t)(tm + 8 * hi) * N + tn + r16;
    float* c10 = c00 + (size_t)16 * N;
    #pragma unroll
    for (int r = 0; r < 8; ++r) {
        c00[(size_t)r * N]      = acc00[r];
        c00[(size_t)r * N + 16] = acc01[r];
        c10[(size_t)r * N]      = acc10[r];
        c10[(size_t)r * N + 16] = acc11[r];
    }
}

// ---------------- additive scores: tanh(q[b,h]+k[k,h]) . wv ----------------
// 16x16 output tile per 256-thread block; q/k tiles staged in LDS (row pad 132
// floats -> bank = h + 4*row mod 64, conflict-free across the 16 rows).
__global__ void scores_kernel(const float* __restrict__ qp, const float* __restrict__ kp,
                              const float* __restrict__ wv, float* __restrict__ scores) {
    __shared__ float sq[16][132];
    __shared__ float sk[16][132];
    __shared__ float sw[HN];
    const int b0 = blockIdx.x << 4;
    const int k0 = blockIdx.y << 4;
    const int t  = threadIdx.x;   // 256
    for (int i = t; i < 16 * HN; i += 256) {
        int r = i >> 7, h = i & (HN - 1);
        sq[r][h] = qp[(size_t)(b0 + r) * HN + h];
        sk[r][h] = kp[(size_t)(k0 + r) * HN + h];
    }
    if (t < HN) sw[t] = wv[t];
    __syncthreads();

    const int bb = t >> 4, kk = t & 15;
    float acc = 0.f;
    #pragma unroll 8
    for (int h = 0; h < HN; ++h) {
        float x  = sq[bb][h] + sk[kk][h];
        float th = __builtin_amdgcn_tanhf(x);   // probe: native v_tanh_f32 (gfx1250 TRANS)
        acc = fmaf(sw[h], th, acc);
    }
    scores[(size_t)(b0 + bb) * KN + k0 + kk] = acc;
}

// ---------------- row softmax over K=2048, emit bf16 probs ----------------
__global__ void softmax_kernel(const float* __restrict__ scores, __bf16* __restrict__ attn) {
    const int row = blockIdx.x;
    const int t   = threadIdx.x;     // 256, wave32 -> 8 waves
    const float* s = scores + (size_t)row * KN;
    __shared__ float red[8];

    float v[8];
    float m = -3.402823466e38f;
    #pragma unroll
    for (int i = 0; i < 8; ++i) { v[i] = s[t + i * 256]; m = fmaxf(m, v[i]); }
    #pragma unroll
    for (int off = 16; off > 0; off >>= 1) m = fmaxf(m, __shfl_xor(m, off, 32));
    if ((t & 31) == 0) red[t >> 5] = m;
    __syncthreads();
    m = red[0];
    #pragma unroll
    for (int i = 1; i < 8; ++i) m = fmaxf(m, red[i]);
    __syncthreads();

    float sum = 0.f;
    #pragma unroll
    for (int i = 0; i < 8; ++i) { v[i] = __expf(v[i] - m); sum += v[i]; }
    #pragma unroll
    for (int off = 16; off > 0; off >>= 1) sum += __shfl_xor(sum, off, 32);
    if ((t & 31) == 0) red[t >> 5] = sum;
    __syncthreads();
    sum = red[0] + red[1] + red[2] + red[3] + red[4] + red[5] + red[6] + red[7];
    const float inv = 1.0f / sum;
    #pragma unroll
    for (int i = 0; i < 8; ++i)
        attn[(size_t)row * KN + t + i * 256] = (__bf16)(v[i] * inv);
}

extern "C" void kernel_launch(void* const* d_in, const int* in_sizes, int n_in,
                              void* d_out, int out_size, void* d_ws, size_t ws_size,
                              hipStream_t stream) {
    const float* queries = (const float*)d_in[0];  // [1024,512]
    const float* keys    = (const float*)d_in[1];  // [2048,512]
    const float* values  = (const float*)d_in[2];  // [2048,512]
    const float* Wq      = (const float*)d_in[3];  // [128,512]
    const float* Wk      = (const float*)d_in[4];  // [128,512]
    const float* wv      = (const float*)d_in[5];  // [128]
    float* out = (float*)d_out;                    // [1024,512]

    // workspace carve-up (all regions 16B-aligned)
    char* ws = (char*)d_ws;
    __bf16* q_bf   = (__bf16*)ws;  ws += (size_t)BN * DN * 2;   // 1 MB
    __bf16* k_bf   = (__bf16*)ws;  ws += (size_t)KN * DN * 2;   // 2 MB
    __bf16* Wq_bf  = (__bf16*)ws;  ws += (size_t)HN * DN * 2;   // 128 KB
    __bf16* Wk_bf  = (__bf16*)ws;  ws += (size_t)HN * DN * 2;   // 128 KB
    __bf16* vT_bf  = (__bf16*)ws;  ws += (size_t)DN * KN * 2;   // 2 MB
    float*  qproj  = (float*)ws;   ws += (size_t)BN * HN * 4;   // 512 KB
    float*  kproj  = (float*)ws;   ws += (size_t)KN * HN * 4;   // 1 MB
    float*  scores = (float*)ws;   ws += (size_t)BN * KN * 4;   // 8 MB
    __bf16* attn   = (__bf16*)ws;  ws += (size_t)BN * KN * 2;   // 4 MB

    // 1) convert inputs to bf16
    cvt_f32_bf16<<<(BN * DN) / 256, 256, 0, stream>>>(queries, q_bf, BN * DN);
    cvt_f32_bf16<<<(KN * DN) / 256, 256, 0, stream>>>(keys,    k_bf, KN * DN);
    cvt_f32_bf16<<<(HN * DN) / 256, 256, 0, stream>>>(Wq,      Wq_bf, HN * DN);
    cvt_f32_bf16<<<(HN * DN) / 256, 256, 0, stream>>>(Wk,      Wk_bf, HN * DN);
    cvt_transpose_vals<<<dim3(KN / 32, DN / 32), 256, 0, stream>>>(values, vT_bf);

    // 2) projections via WMMA (NT GEMM, K contiguous both sides), 32x32 tiles
    gemm_bf16_nt<<<(BN / 32) * (HN / 32) / 8, 256, 0, stream>>>(q_bf, Wq_bf, qproj, BN, HN, DN);
    gemm_bf16_nt<<<(KN / 32) * (HN / 32) / 8, 256, 0, stream>>>(k_bf, Wk_bf, kproj, KN, HN, DN);

    // 3) additive attention scores (VALU/tanh bound — the hot kernel)
    scores_kernel<<<dim3(BN / 16, KN / 16), 256, 0, stream>>>(qproj, kproj, wv, scores);

    // 4) softmax over keys, probs as bf16
    softmax_kernel<<<BN, 256, 0, stream>>>(scores, attn);

    // 5) out = attn @ values via WMMA (vT_bf gives K-contiguous B operand)
    gemm_bf16_nt<<<(BN / 32) * (DN / 32) / 8, 256, 0, stream>>>(attn, vT_bf, out, BN, DN, KN);
}
